// MultiheadAttention_15539191677067
// MI455X (gfx1250) — matrix operand order repaired and checked by reading, not compiled, not run on hardware
//
#include <hip/hip_runtime.h>
#include <hip/hip_bf16.h>

typedef __attribute__((ext_vector_type(16))) _Float16 v16h;
typedef __attribute__((ext_vector_type(8)))  _Float16 v8h;
typedef __attribute__((ext_vector_type(8)))  float    v8f;
typedef __attribute__((ext_vector_type(4)))  float    v4f;

// Problem constants (from reference)
constexpr int BATCH = 4;
constexpr int NHEAD = 16;
constexpr int SEQ   = 2048;
constexpr int HDIM  = 64;
constexpr int EMB   = 1024;                 // NHEAD * HDIM
constexpr float SCALE = 0.03125f;           // 1/sqrt(EMB) = 1/32

// LDS logits strip in f16: 16 rows x SEQ cols per wave.
// Row stride 2056 halves = 4112 bytes -> per-row bank shift of 4 dwords:
// conflict-free b128 accesses across 16 lanes. Per wave: 16*4112 = 65792 B;
// 4 waves = 263168 B (fits CDNA5's 320 KB WGP LDS; one wave per SIMD32).
constexpr int SSTRIDE = 2056;               // in halves

__global__ __launch_bounds__(128)
void mha_wmma_f16(const float* __restrict__ Qp,
                  const float* __restrict__ Kp,
                  const float* __restrict__ Vp,
                  float* __restrict__ outp,
                  float* __restrict__ attnp) {
    extern __shared__ char smem_raw[];
    _Float16* smem = (_Float16*)smem_raw;

    const int tid  = threadIdx.x;
    const int w    = tid >> 5;          // wave id in block (0..3)
    const int lane = tid & 31;
    const int half = lane >> 4;         // 0: lanes 0-15, 1: lanes 16-31
    const int lm   = lane & 15;

    const int bid      = blockIdx.x;
    const int strip_id = bid & 31;          // 32 strips of 64 q-rows per (b,h)
    const int h        = (bid >> 5) & 15;
    const int b        = bid >> 9;
    const int q0       = strip_id * 64 + w * 16;   // this wave's 16 q rows

    _Float16* strip = smem + w * 16 * SSTRIDE;

    // ------------------------------------------------------------------
    // Load Q tile as two f16 A-fragments (16x32 each, covering D=0..63),
    // with the 1/sqrt(E) scale folded in.
    // A fragment as packed here (contiguous k, matched by B): lane M = lane&15 (both halves);
    // halves 0..7 -> K = 16*half + j ; halves 8..15 -> K = 16*half + 8 + j
    // ------------------------------------------------------------------
    v16h aq0, aq1;
    {
        const float* qrow = Qp + ((size_t)(b * SEQ) + q0 + lm) * EMB + h * HDIM;
        const int d0 = 16 * half;
        const int d1 = 16 * half + 8;
        #pragma unroll
        for (int j = 0; j < 8; ++j) {
            aq0[j]     = (_Float16)(qrow[d0 + j]      * SCALE);
            aq0[8 + j] = (_Float16)(qrow[d1 + j]      * SCALE);
            aq1[j]     = (_Float16)(qrow[d0 + 32 + j] * SCALE);
            aq1[8 + j] = (_Float16)(qrow[d1 + 32 + j] * SCALE);
        }
    }

    // ------------------------------------------------------------------
    // Pass 1: logits = (Q*scale) @ K^T  -> LDS strip (f16)
    // B fragment as packed here (same contiguous k as A): lane col n = lane&15; element j -> K = 16*half + j
    // C layout (f32 16x16): reg r -> row r + 8*half, col = lane&15
    // ------------------------------------------------------------------
    for (int n0 = 0; n0 < SEQ; n0 += 16) {
        const float* krow = Kp + ((size_t)(b * SEQ) + n0 + lm) * EMB + h * HDIM;
        const int kb = 16 * half;
        v16h bk0, bk1;
        #pragma unroll
        for (int j = 0; j < 16; ++j) {
            bk0[j] = (_Float16)krow[kb + j];        // D = 0..31
            bk1[j] = (_Float16)krow[kb + 32 + j];   // D = 32..63
        }
        v8f c = {};
        c = __builtin_amdgcn_wmma_f32_16x16x32_f16(false, aq0, false, bk0,
                                                   (short)0, c, false, false);
        c = __builtin_amdgcn_wmma_f32_16x16x32_f16(false, aq1, false, bk1,
                                                   (short)0, c, false, false);
        #pragma unroll
        for (int r = 0; r < 8; ++r)
            strip[(r + 8 * half) * SSTRIDE + n0 + lm] = (_Float16)c[r];
    }
    __syncthreads();

    // ------------------------------------------------------------------
    // Row softmax statistics: lane handles row (lane&15), columns
    // [half*1024, half*1024+1024), b128 LDS reads. Halves combined with
    // wave32 shfl_xor(16).
    // ------------------------------------------------------------------
    float m = -3.0e38f, rs;
    {
        const v8h* rp8 = (const v8h*)(strip + lm * SSTRIDE + half * 1024);
        #pragma unroll 4
        for (int i = 0; i < 128; ++i) {
            v8h x = rp8[i];
            float m0 = fmaxf(fmaxf((float)x[0], (float)x[1]),
                             fmaxf((float)x[2], (float)x[3]));
            float m1 = fmaxf(fmaxf((float)x[4], (float)x[5]),
                             fmaxf((float)x[6], (float)x[7]));
            m = fmaxf(m, fmaxf(m0, m1));
        }
        m = fmaxf(m, __shfl_xor(m, 16, 32));
        float s = 0.0f;
        #pragma unroll 4
        for (int i = 0; i < 128; ++i) {
            v8h x = rp8[i];
            #pragma unroll
            for (int j = 0; j < 8; ++j)
                s += __expf((float)x[j] - m);
        }
        s += __shfl_xor(s, 16, 32);
        rs = 1.0f / s;
    }
    __syncthreads();

    // ------------------------------------------------------------------
    // Pass 2: normalized weights -> global (single write of the 1.07 GB attn
    // tensor, non-temporal b128 stores: keep K/V resident in L2), weights
    // packed to f16 A-fragment, PV via 4 WMMAs per 32-wide K step.
    // ------------------------------------------------------------------
    v8f acc0 = {}, acc1 = {}, acc2 = {}, acc3 = {};
    float* attn_row = attnp +
        ((((size_t)b * NHEAD + h) * SEQ) + q0 + lm) * (size_t)SEQ;
    const _Float16* srow = strip + lm * SSTRIDE;

    for (int k0 = 0; k0 < SEQ; k0 += 32) {
        // weights A-fragment: row = lane&15 (this lane's own softmax stats)
        const int c0 = k0 + 16 * half;
        const int c1 = k0 + 16 * half + 8;
        v8h l0 = *(const v8h*)(srow + c0);
        v8h l1 = *(const v8h*)(srow + c1);
        v16h aw;
        v4f w0a, w0b, w1a, w1b;
        #pragma unroll
        for (int j = 0; j < 4; ++j) {
            float wa = __expf((float)l0[j]     - m) * rs;
            float wb = __expf((float)l0[4 + j] - m) * rs;
            float wc = __expf((float)l1[j]     - m) * rs;
            float wd = __expf((float)l1[4 + j] - m) * rs;
            w0a[j] = wa; w0b[j] = wb; w1a[j] = wc; w1b[j] = wd;
            aw[j]      = (_Float16)wa;
            aw[4 + j]  = (_Float16)wb;
            aw[8 + j]  = (_Float16)wc;
            aw[12 + j] = (_Float16)wd;
        }
        __builtin_nontemporal_store(w0a, (v4f*)(attn_row + c0));
        __builtin_nontemporal_store(w0b, (v4f*)(attn_row + c0 + 4));
        __builtin_nontemporal_store(w1a, (v4f*)(attn_row + c1));
        __builtin_nontemporal_store(w1b, (v4f*)(attn_row + c1 + 4));

        // V B-fragments: lane col d = g*16 + lm; element j -> k = 16*half + j
        const float* vbase = Vp + ((size_t)(b * SEQ) + k0 + 16 * half) * EMB
                               + h * HDIM + lm;
        v16h bv0, bv1, bv2, bv3;
        #pragma unroll
        for (int j = 0; j < 16; ++j) {
            const float* vr = vbase + (size_t)j * EMB;
            bv0[j] = (_Float16)vr[0];
            bv1[j] = (_Float16)vr[16];
            bv2[j] = (_Float16)vr[32];
            bv3[j] = (_Float16)vr[48];
        }
        acc0 = __builtin_amdgcn_wmma_f32_16x16x32_f16(false, aw, false, bv0,
                                                      (short)0, acc0, false, false);
        acc1 = __builtin_amdgcn_wmma_f32_16x16x32_f16(false, aw, false, bv1,
                                                      (short)0, acc1, false, false);
        acc2 = __builtin_amdgcn_wmma_f32_16x16x32_f16(false, aw, false, bv2,
                                                      (short)0, acc2, false, false);
        acc3 = __builtin_amdgcn_wmma_f32_16x16x32_f16(false, aw, false, bv3,
                                                      (short)0, acc3, false, false);
    }

    // Output: C layout row = r + 8*half, col = lane&15 within each d-group of 16
    #pragma unroll
    for (int r = 0; r < 8; ++r) {
        float* orow = outp + ((size_t)(b * SEQ) + q0 + r + 8 * half) * EMB
                           + h * HDIM + lm;
        __builtin_nontemporal_store(acc0[r], orow);
        __builtin_nontemporal_store(acc1[r], orow + 16);
        __builtin_nontemporal_store(acc2[r], orow + 32);
        __builtin_nontemporal_store(acc3[r], orow + 48);
    }
}

extern "C" void kernel_launch(void* const* d_in, const int* in_sizes, int n_in,
                              void* d_out, int out_size, void* d_ws, size_t ws_size,
                              hipStream_t stream) {
    (void)in_sizes; (void)n_in; (void)d_ws; (void)ws_size; (void)out_size;
    const float* q = (const float*)d_in[0];
    const float* k = (const float*)d_in[1];
    const float* v = (const float*)d_in[2];
    float* out  = (float*)d_out;                                   // [B,S,E]
    float* attn = out + (size_t)BATCH * SEQ * EMB;                 // [B,H,S,S]

    const dim3 grid(BATCH * NHEAD * (SEQ / 64));   // 2048 blocks
    const dim3 block(128);                         // 4 wave32 per block
    const size_t shmem = 4u * 16u * SSTRIDE * sizeof(_Float16);    // 263168 B
    mha_wmma_f16<<<grid, block, shmem, stream>>>(q, k, v, out, attn);
}
